// SpectralHead_1382979469658
// MI455X (gfx1250) — matrix operand order repaired and checked by reading, not compiled, not run on hardware
//
#include <hip/hip_runtime.h>
#include <hip/hip_bf16.h>
#include <math.h>

// ---------- CDNA5 (gfx1250) types ----------
typedef __attribute__((ext_vector_type(16))) _Float16 v16h;
typedef __attribute__((ext_vector_type(8)))  _Float16 v8h;
typedef __attribute__((ext_vector_type(8)))  float    v8f;
typedef __attribute__((ext_vector_type(4)))  float    v4f;
typedef __attribute__((ext_vector_type(4)))  unsigned v4u;
typedef __attribute__((ext_vector_type(8)))  int      v8i;
typedef __attribute__((ext_vector_type(4)))  int      v4i;

#define B_  8
#define T_  2048
#define C_  1024
#define H_  64
#define SCALE_ (1.0f/32.0f)   // C^-0.5, sqrt(1024)=32

#if __has_builtin(__builtin_amdgcn_tensor_load_to_lds) && \
    __has_builtin(__builtin_amdgcn_s_wait_tensorcnt)
#define USE_TDM 1
#else
#define USE_TDM 0
#warning "CDNA5 TDM builtins unavailable: proj_kernel uses register-staged LDS copy fallback"
#endif

__device__ __forceinline__ v8f wmma16(v16h a, v16h b, v8f c) {
  // D = A(16x32 f16) x B(32x16 f16) + C(16x16 f32)
  return __builtin_amdgcn_wmma_f32_16x16x32_f16(false, a, false, b, (short)0, c,
                                                false, false);
}

union A16 { v16h v; v8h h8[2]; _Float16 e[16]; unsigned u[8]; };
union H2U { _Float16 h[2]; unsigned u; };

// generic LDS pointer -> 32-bit LDS byte address (low 32 bits of the
// shared-aperture generic address are the LDS offset on CDNA5)
__device__ __forceinline__ unsigned lds_addr_of(const void* p) {
  return (unsigned)(size_t)p;
}

#if USE_TDM
// One TDM descriptor: 2D tile, data_size=8B. Tile = 8 units (64B row chunk)
// x 192 rows; tensor row pitch = 256 units (2048B = 1024 halves).
// Group0: count=1 | lds_addr | global_addr | type=2 (bits 127:126).
__device__ __forceinline__ void tdm_load_slice(const _Float16* gsrc,
                                               unsigned lds_byte_addr) {
  const size_t ga = (size_t)gsrc;
  v4u g0 = { 1u, lds_byte_addr, (unsigned)ga,
             (unsigned)((ga >> 32) & 0x01FFFFFFu) | 0x80000000u };
  // g1: [17:16] data_size=3(8B); tensor_dim0=256; tensor_dim1=192;
  //     tile_dim0=8; tile_dim1=192; tensor_dim0_stride=256.
  v8i g1 = { 0x00030000, 0x01000000, 0x00C00000, (int)(8u << 16),
             192, 256, 0, 0 };
  v4i gz = { 0, 0, 0, 0 };
#if defined(__clang_major__) && (__clang_major__ >= 23)
  v8i gz8 = { 0, 0, 0, 0, 0, 0, 0, 0 };
  __builtin_amdgcn_tensor_load_to_lds(g0, g1, gz, gz, gz8, 0);
#else
  __builtin_amdgcn_tensor_load_to_lds(g0, g1, gz, gz, 0);
#endif
}
#endif

// ============================================================
// Kernel 0: transpose Wq/Wk/Wv (C x H f32) -> Wt[m][h][c] f16
// ============================================================
__global__ __launch_bounds__(256) void wt_kernel(const float* __restrict__ Wq,
                                                 const float* __restrict__ Wk,
                                                 const float* __restrict__ Wv,
                                                 _Float16* __restrict__ Wt) {
  int idx = blockIdx.x * 256 + threadIdx.x;     // 3*64*1024 total
  int m = idx >> 16;
  int r = idx & 65535;
  int h = r >> 10;
  int c = r & 1023;
  const float* W = (m == 0) ? Wq : ((m == 1) ? Wk : Wv);
  Wt[idx] = (_Float16)W[c * H_ + h];
}

// ============================================================
// Kernel 1: fused projection. 4 waves/block, one 16-token tile each,
// sharing double-buffered LDS weight slices (12KB/step) fed by the TDM.
//   qh[b][t][h], kh[b][t][h] (f16), vt[b][h][t] (f16, transposed)
// ============================================================
__global__ __launch_bounds__(128) void proj_kernel(const float* __restrict__ x,
                                                   const _Float16* __restrict__ Wt,
                                                   _Float16* __restrict__ qh,
                                                   _Float16* __restrict__ kh,
                                                   _Float16* __restrict__ vt) {
  // slice layout: [row = m*64+h][32 c-halves], row-major, 12KB per buffer
  __shared__ _Float16 sw[2][192 * 32];

  const int tid = threadIdx.x;
  const int wid = tid >> 5;
  const int gw  = blockIdx.x * 4 + wid;         // 1024 waves
  const int b   = gw >> 7;
  const int tb  = (gw & 127) << 4;
  const int L   = tid & 31;
  const int ln  = L & 15;
  const int hf  = L >> 4;

  const float* xrow = x + ((size_t)b * T_ + tb + ln) * C_;

  v8f acc[3][4];
  #pragma unroll
  for (int m = 0; m < 3; ++m)
    #pragma unroll
    for (int ht = 0; ht < 4; ++ht) acc[m][ht] = v8f{};

#if USE_TDM
  if (tid < 32) tdm_load_slice(Wt, lds_addr_of(&sw[0][0]));   // slice 0
#else
  // fallback prologue: register-staged cooperative copy of slice 0
  {
    #pragma unroll
    for (int i = 0; i < 6; ++i) {
      const int chunk = tid + i * 128;          // 768 x 16B chunks
      const int row = chunk >> 2, cp = chunk & 3;
      v4u t0 = *(const v4u*)(Wt + (size_t)row * C_ + cp * 8);
      *(v4u*)&sw[0][chunk * 8] = t0;
    }
  }
#endif

  for (int kk = 0; kk < C_ / 32; ++kk) {
    const int c0 = kk * 32;
    const int c0n = ((kk + 1) & 31) * 32;       // wraps; extra DMA harmless
    const int cbuf = kk & 1, nbuf = cbuf ^ 1;

    // A operand for this step: x row, K(j) = hf*16 + j
    v4f f0 = *(const v4f*)(xrow + c0 + hf * 16);
    v4f f1 = *(const v4f*)(xrow + c0 + hf * 16 + 4);
    v4f f2 = *(const v4f*)(xrow + c0 + hf * 16 + 8);
    v4f f3 = *(const v4f*)(xrow + c0 + hf * 16 + 12);

#if USE_TDM
    __syncthreads();                            // readers done with nbuf
    if (tid < 32) {
      tdm_load_slice(Wt + c0n, lds_addr_of(&sw[nbuf][0]));
      __builtin_amdgcn_s_wait_tensorcnt(1);     // slice kk complete
    }
    __syncthreads();                            // slice kk visible to all
#else
    // prefetch next slice into registers (overlaps with compute)
    v4u pend[6];
    #pragma unroll
    for (int i = 0; i < 6; ++i) {
      const int chunk = tid + i * 128;
      const int row = chunk >> 2, cp = chunk & 3;
      pend[i] = *(const v4u*)(Wt + (size_t)row * C_ + c0n + cp * 8);
    }
    __syncthreads();                            // slice kk stores visible
#endif

    A16 a;
    #pragma unroll
    for (int i = 0; i < 4; ++i) {
      a.e[i]      = (_Float16)f0[i];
      a.e[4 + i]  = (_Float16)f1[i];
      a.e[8 + i]  = (_Float16)f2[i];
      a.e[12 + i] = (_Float16)f3[i];
    }

    // Batch all 12 B-operand ds_loads, then run 12 WMMAs back-to-back
    v16h bv[12];
    #pragma unroll
    for (int m = 0; m < 3; ++m)
      #pragma unroll
      for (int ht = 0; ht < 4; ++ht) {
        const int row = m * 64 + ht * 16 + ln;
        bv[m * 4 + ht] = *(const v16h*)&sw[cbuf][row * 32 + hf * 16];
      }
    #pragma unroll
    for (int m = 0; m < 3; ++m)
      #pragma unroll
      for (int ht = 0; ht < 4; ++ht)
        acc[m][ht] = wmma16(a.v, bv[m * 4 + ht], acc[m][ht]);

#if !USE_TDM
    __syncthreads();                            // all reads of nbuf done
    #pragma unroll
    for (int i = 0; i < 6; ++i) {
      const int chunk = tid + i * 128;
      *(v4u*)&sw[nbuf][chunk * 8] = pend[i];
    }
#endif
  }

  // Store q, k row-major f16 (C layout: M = r + 8*hf -> token, N = ln -> h)
  #pragma unroll
  for (int m = 0; m < 2; ++m) {
    _Float16* dst = m ? kh : qh;
    #pragma unroll
    for (int ht = 0; ht < 4; ++ht) {
      const int h = ht * 16 + ln;
      #pragma unroll
      for (int r = 0; r < 8; ++r) {
        const int trow = tb + 8 * hf + r;
        dst[((size_t)b * T_ + trow) * H_ + h] = (_Float16)acc[m][ht][r];
      }
    }
  }
  // Store V transposed: vt[b][h][t]; 8 consecutive t per lane -> 16B store
  #pragma unroll
  for (int ht = 0; ht < 4; ++ht) {
    const int h = ht * 16 + ln;
    v8h pk;
    #pragma unroll
    for (int r = 0; r < 8; ++r) pk[r] = (_Float16)acc[2][ht][r];
    *(v8h*)(vt + ((size_t)b * H_ + h) * T_ + tb + 8 * hf) = pk;
  }
}

// ============================================================
// Kernel 2: fused flash attention. One wave -> 16 query rows.
//   S^T = K·Q^T (scores transposed), O^T = V^T·P^T.
//   All of a tile's loads (K, V^T, rbias) are issued up front; the loop is
//   unrolled x2 so tile i+1 loads issue under tile i WMMAs + softmax VALU.
// ============================================================
__global__ __launch_bounds__(128) void attn_kernel(const _Float16* __restrict__ qh,
                                                   const _Float16* __restrict__ kh,
                                                   const _Float16* __restrict__ vt,
                                                   const float* __restrict__ rbias,
                                                   float* __restrict__ out) {
  const int wid = threadIdx.x >> 5;
  const int gw  = blockIdx.x * 4 + wid;         // 1024 waves
  const int b   = gw >> 7;
  const int tb  = (gw & 127) << 4;
  const int L   = threadIdx.x & 31;
  const int ln  = L & 15;
  const int hf  = L >> 4;
  const int t   = tb + ln;                      // this lane's query column

  // Q as B operand (column t, K(j) = hf*16 + j): 32B contiguous per lane
  const _Float16* qrow = qh + ((size_t)b * T_ + t) * H_;
  const v16h qb0 = *(const v16h*)(qrow + hf * 16);        // h 0..31
  const v16h qb1 = *(const v16h*)(qrow + 32 + hf * 16);   // h 32..63

  v8f o[4];
  #pragma unroll
  for (int ht = 0; ht < 4; ++ht) o[ht] = v8f{};
  float m_i = -__builtin_inff();
  float l_i = 0.0f;

  const float*    brow  = rbias + (size_t)t * T_;
  const _Float16* kbase = kh + (size_t)b * T_ * H_;
  const _Float16* vbase = vt + (size_t)b * H_ * T_;

  #pragma unroll 2
  for (int s0 = 0; s0 < tb + 16; s0 += 32) {
    if (s0 + 32 < tb + 16)
      __builtin_prefetch(kbase + (size_t)(s0 + 32 + ln) * H_, 0, 0);

    // ================= issue ALL loads for this tile =================
    const _Float16* k0 = kbase + (size_t)(s0 + ln) * H_;
    const _Float16* k1 = k0 + 16 * H_;
    A16 a00, a01, a10, a11;
    a00.h8[0] = *(const v8h*)(k0 + hf * 16);      a00.h8[1] = *(const v8h*)(k0 + 8 + hf * 16);
    a01.h8[0] = *(const v8h*)(k0 + 32 + hf * 16); a01.h8[1] = *(const v8h*)(k0 + 40 + hf * 16);
    a10.h8[0] = *(const v8h*)(k1 + hf * 16);      a10.h8[1] = *(const v8h*)(k1 + 8 + hf * 16);
    a11.h8[0] = *(const v8h*)(k1 + 32 + hf * 16); a11.h8[1] = *(const v8h*)(k1 + 40 + hf * 16);

    A16 av[4];
    #pragma unroll
    for (int ht = 0; ht < 4; ++ht) {
      const _Float16* vr = vbase + (size_t)(ht * 16 + ln) * T_ + s0;
      av[ht].h8[0] = *(const v8h*)(vr + hf * 16);
      av[ht].h8[1] = *(const v8h*)(vr + 8 + hf * 16);
    }

    v4f rb00 = *(const v4f*)(brow + s0 + hf * 8);
    v4f rb01 = *(const v4f*)(brow + s0 + hf * 8 + 4);
    v4f rb10 = *(const v4f*)(brow + s0 + 16 + hf * 8);
    v4f rb11 = *(const v4f*)(brow + s0 + 16 + hf * 8 + 4);

    // ---- S^T: 2 tiles (keys s0..+15, s0+16..+31), K=64 -> 4 WMMAs ----
    v8f z = v8f{};
    v8f st0 = wmma16(a00.v, qb0, z);  st0 = wmma16(a01.v, qb1, st0);
    v8f st1 = wmma16(a10.v, qb0, z);  st1 = wmma16(a11.v, qb1, st1);

    // ---- bias + analytic mask (allowed := s==t || (s<t && rbias>0)) ----
    float w0[8], w1[8];
    #pragma unroll
    for (int r = 0; r < 8; ++r) {
      const int   sv0 = s0 + hf * 8 + r;        // C layout: M = r + 8*hf
      const int   sv1 = sv0 + 16;
      const float rb0 = (r < 4) ? rb00[r] : rb01[r - 4];
      const float rb1 = (r < 4) ? rb10[r] : rb11[r - 4];
      const bool ok0 = (sv0 == t) || ((sv0 < t) && (rb0 > 0.0f));
      const bool ok1 = (sv1 == t) || ((sv1 < t) && (rb1 > 0.0f));
      w0[r] = ok0 ? st0[r] * SCALE_ + rb0 : -1e30f;
      w1[r] = ok1 ? st1[r] * SCALE_ + rb1 : -1e30f;
    }

    // ---- online softmax: 16 in-lane values + one xor-16 shuffle ----
    float mx = w0[0];
    #pragma unroll
    for (int r = 1; r < 8; ++r) mx = fmaxf(mx, w0[r]);
    #pragma unroll
    for (int r = 0; r < 8; ++r) mx = fmaxf(mx, w1[r]);
    mx = fmaxf(mx, __shfl_xor(mx, 16));
    const float m_new = fmaxf(m_i, mx);
    const float resc  = __expf(m_i - m_new);    // first iter: exp(-inf)=0
    float p0[8], p1[8], ssum = 0.0f;
    #pragma unroll
    for (int r = 0; r < 8; ++r) {
      p0[r] = __expf(w0[r] - m_new);
      p1[r] = __expf(w1[r] - m_new);
      ssum += p0[r] + p1[r];
    }
    ssum += __shfl_xor(ssum, 16);
    l_i = l_i * resc + ssum;
    m_i = m_new;
    #pragma unroll
    for (int ht = 0; ht < 4; ++ht) o[ht] = o[ht] * resc;

    // ---- relayout P (C layout of S^T) -> B operand of P^T ----
    // B needs k=s offset = hf*16 + j; own lane covers {8hf..8hf+7, 16+8hf..},
    // partner (xor 16) covers the other 8 — pack to half2, one shuffle each.
    unsigned pk0[4], pk1[4];
    #pragma unroll
    for (int i = 0; i < 4; ++i) {
      H2U u0; u0.h[0] = (_Float16)p0[2 * i]; u0.h[1] = (_Float16)p0[2 * i + 1];
      H2U u1; u1.h[0] = (_Float16)p1[2 * i]; u1.h[1] = (_Float16)p1[2 * i + 1];
      pk0[i] = u0.u; pk1[i] = u1.u;
    }
    A16 pb;
    #pragma unroll
    for (int i = 0; i < 4; ++i) {
      const unsigned r0 = __shfl_xor(pk0[i], 16);
      const unsigned r1 = __shfl_xor(pk1[i], 16);
      pb.u[i]     = hf ? r1     : pk0[i];
      pb.u[4 + i] = hf ? pk1[i] : r0;
    }

    // ---- O^T += V^T(16h x 32s) @ P^T(32s x 16t): 4 WMMAs ----
    #pragma unroll
    for (int ht = 0; ht < 4; ++ht) o[ht] = wmma16(av[ht].v, pb.v, o[ht]);
  }

  // ---- normalize and store: lane owns query t, h = ht*16 + 8*hf + r ----
  const float inv = 1.0f / l_i;
  float* orow = out + ((size_t)b * T_ + t) * H_ + hf * 8;
  #pragma unroll
  for (int ht = 0; ht < 4; ++ht) {
    v4f lo, hi;
    #pragma unroll
    for (int r = 0; r < 4; ++r) {
      lo[r] = o[ht][r] * inv;
      hi[r] = o[ht][4 + r] * inv;
    }
    *(v4f*)(orow + ht * 16)     = lo;
    *(v4f*)(orow + ht * 16 + 4) = hi;
  }
}

// ============================================================
extern "C" void kernel_launch(void* const* d_in, const int* in_sizes, int n_in,
                              void* d_out, int out_size, void* d_ws, size_t ws_size,
                              hipStream_t stream) {
  const float* x     = (const float*)d_in[0];
  const float* Wq    = (const float*)d_in[1];
  const float* Wk    = (const float*)d_in[2];
  const float* Wv    = (const float*)d_in[3];
  const float* rbias = (const float*)d_in[4];
  // d_in[5] = allowed: derived analytically from rbias, not read.
  float* out = (float*)d_out;

  _Float16* ws = (_Float16*)d_ws;
  const size_t WT_OFF = 0;                          // 3*64*1024
  const size_t QH_OFF = WT_OFF + (size_t)3 * H_ * C_;
  const size_t KH_OFF = QH_OFF + (size_t)B_ * T_ * H_;
  const size_t VT_OFF = KH_OFF + (size_t)B_ * T_ * H_;
  _Float16* Wt = ws + WT_OFF;
  _Float16* qh = ws + QH_OFF;
  _Float16* kh = ws + KH_OFF;
  _Float16* vt = ws + VT_OFF;

  wt_kernel<<<(3 * H_ * C_) / 256, 256, 0, stream>>>(Wq, Wk, Wv, Wt);
  // 1024 waves = 8 batches * 128 token tiles; 4 waves/block
  proj_kernel<<<256, 128, 0, stream>>>(x, Wt, qh, kh, vt);
  attn_kernel<<<256, 128, 0, stream>>>(qh, kh, vt, rbias, out);
}
